// SpGAT_81552839016625
// MI455X (gfx1250) — compile-verified, matrix operand order checked by reading, not run on hardware
//
#include <hip/hip_runtime.h>
#include <hip/hip_bf16.h>

// ---------------------------------------------------------------------------
// Problem constants (from reference)
// ---------------------------------------------------------------------------
#define N_NODES 20000
#define NFEAT   128
#define C1      256      // NHEADS*NHID, also layer-2 width
#define N_REL   474
#define E1      300000
#define E2      60000
#define ALPHA_LRELU 0.2f

// d_out layout (floats): x [N*256] | out_relation_1 [474*256] | ortho [1]
#define OUT_X_ELEMS   (N_NODES * C1)          // 5,120,000
#define OUT_REL_ELEMS (N_REL * C1)            // 121,344
#define OUT_ORTHO_OFF (OUT_X_ELEMS + OUT_REL_ELEMS)

// GEMM edge-chunk for edge_embed @ a_rel^T (keeps 16MB tile L2-hot)
#define CHUNK 16384

// ---------------------------------------------------------------------------
// WMMA types
// ---------------------------------------------------------------------------
typedef __attribute__((ext_vector_type(16))) __bf16        v16bf;
typedef __attribute__((ext_vector_type(8)))  float         v8f;
typedef __attribute__((ext_vector_type(8)))  unsigned short u16x8;
typedef __attribute__((ext_vector_type(2)))  unsigned int   u32x2;

union FragBF {
    u16x8 h[2];
    v16bf v;
};

// True packed f32x2 -> bf16x2: one v_cvt_pk_bf16_f32 (mnemonic verified in this
// toolchain's own gfx1250 output; dst.lo = cvt(src0), dst.hi = cvt(src1)).
__device__ __forceinline__ unsigned int pk2bf(float lo, float hi) {
    unsigned int r;
    asm("v_cvt_pk_bf16_f32 %0, %1, %2" : "=v"(r) : "v"(lo), "v"(hi));
    return r;
}
__device__ __forceinline__ unsigned short f2bf(float x) {
    union { __hip_bfloat16 h; unsigned short u; } cv;
    cv.h = __float2bfloat16(x);
    return cv.u;
}

// ---------------------------------------------------------------------------
// Tiled bf16 WMMA GEMM:  C[M x N] = A[M x K](f32) * op(B)(f32->bf16)
//   transB=1: B is [N x K] row-major (A @ B^T)   (float4-vectorized staging)
//   transB=0: B is [K x N] row-major              (scalar staging, tiny use)
// Block: 128 threads = 4 waves; block tile 128x64; K step 32.
// Wave w owns rows [w*32, w*32+32): 2 A-fragments x 4 B-tiles = 8 WMMAs / K-step.
// Software-pipelined: next tile's global loads are issued into registers while
// the current tile is computed out of LDS, so HBM/L2 latency hides under WMMA.
// All lda/ldb/boff call sites are multiples of 4 floats -> float4 loads legal.
// ---------------------------------------------------------------------------
#define TM 128
#define TN 64
#define TK 32
#define TKP 48   // LDS row stride in ushorts (96B: multiple of 16B for b128 loads)

__global__ __launch_bounds__(128) void gemm_bf16_wmma(
    const float* __restrict__ A, const float* __restrict__ B, float* __restrict__ C,
    int M, int N, int K, int lda, int ldb, int ldc, int boff, int transB)
{
    __shared__ __align__(16) unsigned short As[TM][TKP];
    __shared__ __align__(16) unsigned short Bs[TN][TKP];

    const int tid  = threadIdx.x;
    const int lane = tid & 31;
    const int wv   = tid >> 5;
    const int m0   = blockIdx.y * TM;
    const int n0   = blockIdx.x * TN;

    v8f acc0[4] = {};   // rows [wv*32, wv*32+16)
    v8f acc1[4] = {};   // rows [wv*32+16, wv*32+32)

    const int rbase = wv * 32 + (lane & 15);
    const int kb    = (lane >> 4) * 8;          // half-wave K base per WMMA bf16 layout

    // staging coordinates: thread covers 4 consecutive k (NN path: 16 n, stride 4, at one k)
    const int sr = tid >> 3;                    // 0..15
    const int sk = (tid & 7) * 4;               // 0,4,...,28

    float4 ra[8];        // A tile: 8 float4 per thread
    float4 rb[4];        // B tile (transB): 4 float4 per thread
    float  rbs[16];      // B tile (NN): 16 scalars per thread

    // ---- phase helpers (kept as plain code blocks via macros of the loop) ----
    #define LOAD_TILE(k0)                                                          \
        {                                                                          \
            _Pragma("unroll")                                                      \
            for (int i = 0; i < 8; ++i) {                                          \
                int gr  = m0 + i * 16 + sr;                                        \
                int grc = gr < M ? gr : (M - 1);    /* clamp: address valid */     \
                ra[i] = *(const float4*)(A + (size_t)grc * lda + ((k0) + sk));     \
            }                                                                      \
            if (transB) {                                                          \
                _Pragma("unroll")                                                  \
                for (int i = 0; i < 4; ++i) {                                      \
                    int n = i * 16 + sr;                                           \
                    rb[i] = *(const float4*)(B + (size_t)(n0 + n) * ldb + boff + ((k0) + sk)); \
                }                                                                  \
            } else {                                                               \
                _Pragma("unroll")                                                  \
                for (int i = 0; i < 16; ++i) {                                     \
                    int idx = i * 128 + tid;                                       \
                    int n = idx >> 5, k = idx & 31;                                \
                    rbs[i] = B[(size_t)((k0) + k) * ldb + boff + (n0 + n)];        \
                }                                                                  \
            }                                                                      \
        }

    #define STORE_TILE()                                                           \
        {                                                                          \
            _Pragma("unroll")                                                      \
            for (int i = 0; i < 8; ++i) {                                          \
                int r  = i * 16 + sr;                                              \
                float4 v = ra[i];                                                  \
                if (m0 + r >= M) { v.x = 0.f; v.y = 0.f; v.z = 0.f; v.w = 0.f; }   \
                u32x2 p; p.x = pk2bf(v.x, v.y); p.y = pk2bf(v.z, v.w);             \
                *(u32x2*)&As[r][sk] = p;                                           \
            }                                                                      \
            if (transB) {                                                          \
                _Pragma("unroll")                                                  \
                for (int i = 0; i < 4; ++i) {                                      \
                    int n = i * 16 + sr;                                           \
                    u32x2 p; p.x = pk2bf(rb[i].x, rb[i].y);                        \
                    p.y = pk2bf(rb[i].z, rb[i].w);                                 \
                    *(u32x2*)&Bs[n][sk] = p;                                       \
                }                                                                  \
            } else {                                                               \
                _Pragma("unroll")                                                  \
                for (int i = 0; i < 16; ++i) {                                     \
                    int idx = i * 128 + tid;                                       \
                    Bs[idx >> 5][idx & 31] = f2bf(rbs[i]);                         \
                }                                                                  \
            }                                                                      \
        }

    // prologue: tile 0 into LDS
    LOAD_TILE(0)
    STORE_TILE()
    __syncthreads();

    for (int k0 = 0; k0 < K; k0 += TK) {
        const bool more = (k0 + TK) < K;
        if (more) LOAD_TILE(k0 + TK)   // in-flight during compute below

        // ---- compute current tile from LDS ----
        FragBF fa0, fa1;
        fa0.h[0] = *(const u16x8*)&As[rbase][kb];
        fa0.h[1] = *(const u16x8*)&As[rbase][16 + kb];
        fa1.h[0] = *(const u16x8*)&As[rbase + 16][kb];
        fa1.h[1] = *(const u16x8*)&As[rbase + 16][16 + kb];

        #pragma unroll
        for (int t = 0; t < 4; ++t) {
            FragBF fb;
            int colB = t * 16 + (lane & 15);
            fb.h[0] = *(const u16x8*)&Bs[colB][kb];
            fb.h[1] = *(const u16x8*)&Bs[colB][16 + kb];
            acc0[t] = __builtin_amdgcn_wmma_f32_16x16x32_bf16(
                false, fa0.v, false, fb.v, (short)0, acc0[t], false, false);
            acc1[t] = __builtin_amdgcn_wmma_f32_16x16x32_bf16(
                false, fa1.v, false, fb.v, (short)0, acc1[t], false, false);
        }

        if (more) {
            __syncthreads();           // all waves done reading LDS
            STORE_TILE()
            __syncthreads();           // stores visible before next compute
        }
    }
    #undef LOAD_TILE
    #undef STORE_TILE

    // ---- store: D[M=r+8*hi][N=lane&15] in VGPR r of lane ----
    const int hi = lane >> 4;
    #pragma unroll
    for (int t = 0; t < 4; ++t) {
        int col = n0 + t * 16 + (lane & 15);
        #pragma unroll
        for (int r = 0; r < 8; ++r) {
            int row0 = m0 + wv * 32 + r + 8 * hi;
            if (row0 < M) C[(size_t)row0 * ldc + col] = acc0[t][r];
            int row1 = row0 + 16;
            if (row1 < M) C[(size_t)row1 * ldc + col] = acc1[t][r];
        }
    }
}

// ---------------------------------------------------------------------------
// Edge pass: one wave per edge; 256 channels -> 8 per lane (two float4 gathers
// per table row; rows are 1KB and L2-resident).
//   m[c]   = hs[src][c] + hd[dst][c] + rel[c]
//   w_h    = exp(-leaky_relu(dot(m_head, a2_head)))   (per head)
//   rowsum[src*nHeads+h] += w_h ;  hprime[src][c] += w_h * m[c]
// rel term: mode 0 = stream (precomputed GEMM chunk), 1 = table[type[e]],
//           2 = table[t0]+table[t1] (2-hop).
// ---------------------------------------------------------------------------
__global__ __launch_bounds__(256) void edge_pass(
    const int* __restrict__ srcIdx, const int* __restrict__ dstIdx,
    const float* __restrict__ hs, const float* __restrict__ hd,
    const float* __restrict__ a2,
    const float* __restrict__ relStream,
    const int* __restrict__ relTypes, const float* __restrict__ relTable,
    int mode, int nEdges, int nHeads,
    float* __restrict__ rowsum, float* __restrict__ hprime)
{
    const int gid  = blockIdx.x * blockDim.x + threadIdx.x;
    const int e    = gid >> 5;
    const int lane = gid & 31;
    if (e >= nEdges) return;

    const int s  = srcIdx[e];
    const int dn = dstIdx[e];
    const int c0 = lane * 8;

    float4 m04, m44;
    {
        const float4* hs4 = (const float4*)(hs + (size_t)s  * C1 + c0);
        const float4* hd4 = (const float4*)(hd + (size_t)dn * C1 + c0);
        float4 r0, r1;
        if (mode == 0) {
            const float4* rp = (const float4*)(relStream + (size_t)e * C1 + c0);
            if (e + 64 < nEdges) __builtin_prefetch(relStream + (size_t)(e + 64) * C1 + c0, 0, 1);
            r0 = rp[0]; r1 = rp[1];
        } else if (mode == 1) {
            const float4* rp = (const float4*)(relTable + (size_t)relTypes[e] * C1 + c0);
            r0 = rp[0]; r1 = rp[1];
        } else {
            const float4* p0 = (const float4*)(relTable + (size_t)relTypes[2 * e]     * C1 + c0);
            const float4* p1 = (const float4*)(relTable + (size_t)relTypes[2 * e + 1] * C1 + c0);
            float4 a0 = p0[0], a1 = p0[1], b0 = p1[0], b1 = p1[1];
            r0.x = a0.x + b0.x; r0.y = a0.y + b0.y; r0.z = a0.z + b0.z; r0.w = a0.w + b0.w;
            r1.x = a1.x + b1.x; r1.y = a1.y + b1.y; r1.z = a1.z + b1.z; r1.w = a1.w + b1.w;
        }
        float4 s0 = hs4[0], s1 = hs4[1], d0 = hd4[0], d1 = hd4[1];
        m04.x = s0.x + d0.x + r0.x; m04.y = s0.y + d0.y + r0.y;
        m04.z = s0.z + d0.z + r0.z; m04.w = s0.w + d0.w + r0.w;
        m44.x = s1.x + d1.x + r1.x; m44.y = s1.y + d1.y + r1.y;
        m44.z = s1.z + d1.z + r1.z; m44.w = s1.w + d1.w + r1.w;
    }

    const float4 a20 = *(const float4*)(a2 + c0);
    const float4 a24 = *(const float4*)(a2 + c0 + 4);
    float d = m04.x * a20.x + m04.y * a20.y + m04.z * a20.z + m04.w * a20.w
            + m44.x * a24.x + m44.y * a24.y + m44.z * a24.z + m44.w * a24.w;

    // reduce dot within the lanes covering one head (wave32)
    const int lph = 32 / nHeads;           // 16 (layer1) or 32 (layer2)
    for (int off = 1; off < lph; off <<= 1) d += __shfl_xor(d, off, lph);

    const float lr = d > 0.0f ? d : ALPHA_LRELU * d;   // leaky_relu
    const float w  = __expf(-lr);

    if ((lane & (lph - 1)) == 0) {
        int head = lane / lph;
        atomicAdd(&rowsum[(size_t)s * nHeads + head], w);
    }
    float* hp = hprime + (size_t)s * C1 + c0;
    atomicAdd(hp + 0, w * m04.x); atomicAdd(hp + 1, w * m04.y);
    atomicAdd(hp + 2, w * m04.z); atomicAdd(hp + 3, w * m04.w);
    atomicAdd(hp + 4, w * m44.x); atomicAdd(hp + 5, w * m44.y);
    atomicAdd(hp + 6, w * m44.z); atomicAdd(hp + 7, w * m44.w);
}

// ---------------------------------------------------------------------------
// Finalize: h /= rowsum (0 -> 1e-12), optional ELU; in place; float4 per thread.
// ---------------------------------------------------------------------------
__global__ __launch_bounds__(256) void finalize_kernel(
    float* __restrict__ h, const float* __restrict__ rowsum,
    int n_quads, int nHeads, int doElu)
{
    int q = blockIdx.x * blockDim.x + threadIdx.x;
    if (q >= n_quads) return;
    int idx = q * 4;
    int n = idx / C1, c = idx % C1;              // c..c+3 share one head
    int head = c / (C1 / nHeads);
    float rs = rowsum[(size_t)n * nHeads + head];
    if (rs == 0.0f) rs = 1e-12f;
    float inv = 1.0f / rs;
    float4 v = *(float4*)(h + idx);
    v.x *= inv; v.y *= inv; v.z *= inv; v.w *= inv;
    if (doElu) {
        v.x = v.x > 0.f ? v.x : (__expf(v.x) - 1.f);
        v.y = v.y > 0.f ? v.y : (__expf(v.y) - 1.f);
        v.z = v.z > 0.f ? v.z : (__expf(v.z) - 1.f);
        v.w = v.w > 0.f ? v.w : (__expf(v.w) - 1.f);
    }
    *(float4*)(h + idx) = v;
}

// ---------------------------------------------------------------------------
// Ortho loss: one wave per Gram entry.
//   att_a heads: 2 x (128x384), 4 groups of 32 rows -> 2*4*32*32 = 8192 entries
//   out_a:          (256x768),  4 groups of 64 rows ->   4*64*64 = 16384 entries
// loss += 0.01 * (dot(row_i,row_j) - delta_ij)^2  (atomic scalar)
// ---------------------------------------------------------------------------
__global__ __launch_bounds__(256) void ortho_kernel(
    const float* __restrict__ att_a, const float* __restrict__ out_a,
    float* __restrict__ loss)
{
    const int gid  = blockIdx.x * blockDim.x + threadIdx.x;
    int w    = gid >> 5;
    const int lane = gid & 31;
    const int total = 8192 + 16384;
    if (w >= total) return;

    const float* base; int g, i, j, rows, d;
    if (w < 8192) {
        int h = w >> 12; w &= 4095;
        g = w >> 10; i = (w >> 5) & 31; j = w & 31;
        base = att_a + (size_t)h * 128 * 384;
        rows = 32; d = 384;
    } else {
        w -= 8192;
        g = w >> 12; i = (w >> 6) & 63; j = w & 63;
        base = out_a; rows = 64; d = 768;
    }
    const float* ri = base + (size_t)(g * rows + i) * d;
    const float* rj = base + (size_t)(g * rows + j) * d;
    float s = 0.0f;
    for (int k = lane; k < d; k += 32) s += ri[k] * rj[k];
    for (int off = 1; off < 32; off <<= 1) s += __shfl_xor(s, off, 32);
    if (lane == 0) {
        float diff = s - (i == j ? 1.0f : 0.0f);
        atomicAdd(loss, 0.01f * diff * diff);
    }
}

// ---------------------------------------------------------------------------
// Host orchestration
// ---------------------------------------------------------------------------
extern "C" void kernel_launch(void* const* d_in, const int* in_sizes, int n_in,
                              void* d_out, int out_size, void* d_ws, size_t ws_size,
                              hipStream_t stream) {
    const float* ent       = (const float*)d_in[0];   // 20000 x 128
    const float* rel       = (const float*)d_in[1];   // 474 x 128
    const float* edge_emb  = (const float*)d_in[2];   // 300000 x 128
    const float* att_a     = (const float*)d_in[3];   // 2 x 128 x 384
    const float* att_a2    = (const float*)d_in[4];   // 2 x 128 (contiguous 256)
    const float* out_a     = (const float*)d_in[5];   // 256 x 768
    const float* out_a2    = (const float*)d_in[6];   // 256
    const float* Wmat      = (const float*)d_in[7];   // 128 x 256
    const int*   edge_list = (const int*)d_in[8];     // [2][300000]
    const int*   edge_type = (const int*)d_in[9];     // 300000
    const int*   el_nhop   = (const int*)d_in[10];    // [2][60000]
    const int*   et_nhop   = (const int*)d_in[11];    // 60000 x 2

    float* out    = (float*)d_out;
    float* outrel = out + OUT_X_ELEMS;                // out_relation_1
    float* ortho  = out + OUT_ORTHO_OFF;              // scalar

    // workspace carve-up (floats)
    float* ws   = (float*)d_ws;
    float* hs1  = ws;                                 // 20000x256
    float* hd1  = hs1 + OUT_X_ELEMS;                  // 20000x256
    float* hp1  = hd1 + OUT_X_ELEMS;                  // layer-1 accum -> becomes x2
    float* hs2  = hp1 + OUT_X_ELEMS;                  // 20000x256
    float* hd2  = hs2 + OUT_X_ELEMS;                  // 20000x256
    float* erch = hd2 + OUT_X_ELEMS;                  // CHUNK x 256 rel-proj chunk
    float* rrl1 = erch + (size_t)CHUNK * C1;          // 474x256: rel @ a_rel1^T
    float* tab2 = rrl1 + OUT_REL_ELEMS;               // 474x256: outrel @ a_rel2^T
    float* rs1  = tab2 + OUT_REL_ELEMS;               // 20000x2
    float* rs2  = rs1 + 2 * N_NODES;                  // 20000

    // zero accumulators (async memset is graph-capturable)
    hipMemsetAsync(hp1, 0, (size_t)OUT_X_ELEMS * 4, stream);
    hipMemsetAsync(rs1, 0, (size_t)2 * N_NODES * 4, stream);
    hipMemsetAsync(rs2, 0, (size_t)N_NODES * 4, stream);
    hipMemsetAsync(out, 0, (size_t)OUT_X_ELEMS * 4, stream);   // layer-2 accum target
    hipMemsetAsync(ortho, 0, 4, stream);

    const dim3 blk(128);
    const dim3 gNode(C1 / TN, (N_NODES + TM - 1) / TM);        // (4, 157)
    const dim3 gRel(C1 / TN, (N_REL + TM - 1) / TM);           // (4, 4)

    // ---- node / table projections (bf16 WMMA GEMMs) ----
    // hs1 = x @ a_src^T (heads fused: att_a viewed as 256x384, cols 0:128)
    gemm_bf16_wmma<<<gNode, blk, 0, stream>>>(ent, att_a, hs1, N_NODES, C1, NFEAT, NFEAT, 384, C1, 0, 1);
    // hd1 = x @ a_dst^T (cols 128:256)
    gemm_bf16_wmma<<<gNode, blk, 0, stream>>>(ent, att_a, hd1, N_NODES, C1, NFEAT, NFEAT, 384, C1, 128, 1);
    // rrl1 = relation_embed @ a_rel1^T (cols 256:384) — feeds 2-hop rel term
    gemm_bf16_wmma<<<gRel, blk, 0, stream>>>(rel, att_a, rrl1, N_REL, C1, NFEAT, NFEAT, 384, C1, 256, 1);
    // out_relation_1 = relation_embed @ W   (NN)
    gemm_bf16_wmma<<<gRel, blk, 0, stream>>>(rel, Wmat, outrel, N_REL, C1, NFEAT, NFEAT, C1, C1, 0, 0);
    // tab2 = out_relation_1 @ a_rel2^T (out_a cols 512:768) — layer-2 rel table
    gemm_bf16_wmma<<<gRel, blk, 0, stream>>>(outrel, out_a, tab2, N_REL, C1, C1, C1, 768, C1, 512, 1);

    const int* el0 = edge_list;            // src row
    const int* el1 = edge_list + E1;       // dst row
    const int* en0 = el_nhop;
    const int* en1 = el_nhop + E2;

    // ---- layer 1: chunked rel GEMM fused (via L2) with edge pass ----
    for (int c0 = 0; c0 < E1; c0 += CHUNK) {
        int mc = E1 - c0; if (mc > CHUNK) mc = CHUNK;
        dim3 gC(C1 / TN, (mc + TM - 1) / TM);
        gemm_bf16_wmma<<<gC, blk, 0, stream>>>(edge_emb + (size_t)c0 * NFEAT, att_a, erch,
                                               mc, C1, NFEAT, NFEAT, 384, C1, 256, 1);
        dim3 gE(((size_t)mc * 32 + 255) / 256);
        edge_pass<<<gE, 256, 0, stream>>>(el0 + c0, el1 + c0, hs1, hd1, att_a2,
                                          erch, nullptr, nullptr, 0, mc, 2, rs1, hp1);
    }
    // layer 1, 2-hop edges: rel term from table sum
    {
        dim3 gE(((size_t)E2 * 32 + 255) / 256);
        edge_pass<<<gE, 256, 0, stream>>>(en0, en1, hs1, hd1, att_a2,
                                          nullptr, et_nhop, rrl1, 2, E2, 2, rs1, hp1);
    }
    // finalize layer 1 (divide + ELU) in place -> hp1 is now x2
    {
        dim3 gF((OUT_X_ELEMS / 4 + 255) / 256);
        finalize_kernel<<<gF, 256, 0, stream>>>(hp1, rs1, OUT_X_ELEMS / 4, 2, 1);
    }

    // ---- layer 2 projections ----
    gemm_bf16_wmma<<<gNode, blk, 0, stream>>>(hp1, out_a, hs2, N_NODES, C1, C1, C1, 768, C1, 0, 1);
    gemm_bf16_wmma<<<gNode, blk, 0, stream>>>(hp1, out_a, hd2, N_NODES, C1, C1, C1, 768, C1, 256, 1);

    // ---- layer 2 edge passes (rel terms are all table lookups) ----
    {
        dim3 gE(((size_t)E1 * 32 + 255) / 256);
        edge_pass<<<gE, 256, 0, stream>>>(el0, el1, hs2, hd2, out_a2,
                                          nullptr, edge_type, tab2, 1, E1, 1, rs2, out);
        dim3 gE2(((size_t)E2 * 32 + 255) / 256);
        edge_pass<<<gE2, 256, 0, stream>>>(en0, en1, hs2, hd2, out_a2,
                                           nullptr, et_nhop, tab2, 2, E2, 1, rs2, out);
        dim3 gF((OUT_X_ELEMS / 4 + 255) / 256);
        finalize_kernel<<<gF, 256, 0, stream>>>(out, rs2, OUT_X_ELEMS / 4, 1, 0);
    }

    // ---- ortho regularizer ----
    {
        dim3 gO(((8192 + 16384) * 32 + 255) / 256);
        ortho_kernel<<<gO, 256, 0, stream>>>(att_a, out_a, ortho);
    }
}
